// fftCorrelation_61443802137052
// MI455X (gfx1250) — compile-verified, matrix operand order checked by reading, not run on hardware
//
#include <hip/hip_runtime.h>
#include <hip/hip_bf16.h>
#include <math.h>

// ---------------------------------------------------------------------------
// CDNA5 (gfx1250) implementation. wave32, WMMA 16x16x32 f16->f32.
// GEMMs: LDS-staged tiles, b128 global traffic, b128 LDS fragment loads.
// ---------------------------------------------------------------------------

typedef __attribute__((ext_vector_type(16))) _Float16 v16h;
typedef __attribute__((ext_vector_type(8)))  _Float16 v8h;
typedef __attribute__((ext_vector_type(8)))  float    v8f;

#define NIMG 8
#define NE   32
#define C3   192   // 3*dim
#define C6   384   // 6*dim
#define C4   256   // 4*dim
#define HW   1024  // 32*32

// ---------------- K1: PatchEmbed (3 stride-2 convs, concat) ----------------
__global__ __launch_bounds__(256) void patch_embed_kernel(
    const float* __restrict__ f1, const float* __restrict__ sw,
    const float* __restrict__ sb, const float* __restrict__ mw,
    const float* __restrict__ mb, const float* __restrict__ lw,
    const float* __restrict__ lb, float* __restrict__ f1p) {
  const int idx = blockIdx.x * 256 + threadIdx.x;
  if (idx >= NIMG * C3 * HW) return;
  const int pix = idx & 1023;
  const int oc  = (idx >> 10) % C3;
  const int img = idx / (C3 * HW);
  const int oy = pix >> 5, ox = pix & 31;
  const float* in = f1 + (size_t)img * 64 * 64 * 64;
  float acc;
  if (oc < 64) {  // 2x2, stride 2, pad 0
    acc = sb[oc];
    const float* w = sw + (size_t)oc * 64 * 4;
    for (int ic = 0; ic < 64; ++ic) {
      const float* ip = in + (size_t)ic * 4096 + (oy * 2) * 64 + ox * 2;
      const float* wp = w + ic * 4;
      acc += ip[0] * wp[0] + ip[1] * wp[1] + ip[64] * wp[2] + ip[65] * wp[3];
    }
  } else {  // 3x3, stride 2, pad 1 (mid / large branch)
    int o2 = oc - 64;
    const float* w;
    if (o2 < 64) { w = mw + (size_t)o2 * 64 * 9; acc = mb[o2]; }
    else         { o2 -= 64; w = lw + (size_t)o2 * 64 * 9; acc = lb[o2]; }
    for (int ic = 0; ic < 64; ++ic) {
      const float* ip = in + (size_t)ic * 4096;
      const float* wp = w + ic * 9;
#pragma unroll
      for (int ky = 0; ky < 3; ++ky) {
        const int iy = oy * 2 - 1 + ky;
        if ((unsigned)iy >= 64u) continue;
#pragma unroll
        for (int kx = 0; kx < 3; ++kx) {
          const int ix = ox * 2 - 1 + kx;
          if ((unsigned)ix >= 64u) continue;
          acc += ip[iy * 64 + ix] * wp[ky * 3 + kx];
        }
      }
    }
  }
  f1p[idx] = acc;
}

// ---------------- K2: Embedd_d (conv s2 + BN(eval) + tanh) -----------------
__global__ __launch_bounds__(256) void embedd_kernel(
    const float* __restrict__ de, const float* __restrict__ ew,
    const float* __restrict__ eb, const float* __restrict__ g,
    const float* __restrict__ b, const float* __restrict__ rm,
    const float* __restrict__ rv, float* __restrict__ d8) {
  const int idx = blockIdx.x * 256 + threadIdx.x;
  if (idx >= NIMG * C6 * HW) return;
  const int pix = idx & 1023;
  const int oc  = (idx >> 10) % C6;
  const int img = idx / (C6 * HW);
  const int oy = pix >> 5, ox = pix & 31;
  const float* in = de + (size_t)img * 128 * 4096;
  const float* w  = ew + (size_t)oc * 128 * 4;
  float acc = eb[oc];
  for (int ic = 0; ic < 128; ++ic) {
    const float* ip = in + (size_t)ic * 4096 + (oy * 2) * 64 + ox * 2;
    const float* wp = w + ic * 4;
    acc += ip[0] * wp[0] + ip[1] * wp[1] + ip[64] * wp[2] + ip[65] * wp[3];
  }
  const float nrm = (acc - rm[oc]) * rsqrtf(rv[oc] + 1e-5f);
  d8[idx] = tanhf(nrm * g[oc] + b[oc]);
}

// ---------------- LDS-staged WMMA GEMM tile (16M x 128N per block) ---------
// Block = 256 threads = 8 waves; wave w owns N columns [w*16, w*16+16).
// Per 32-wide K chunk:
//   - A (16x32 f32) -> f16 As[16][32]           (coalesced b128 loads)
//   - B (32x128 f32) -> f16 BsT[128][40] (K-transposed, 16B-aligned rows)
//   - fragments = two 16-byte ds loads each (consecutive-K runs)
// C tile is staged through LDS for coalesced b128 stores.
__device__ void gemm_tile_block(const float* __restrict__ A,
                                const float* __restrict__ B,
                                float* __restrict__ C, int K, int m0,
                                int nblk) {
  __shared__ _Float16 As[16][32];    //  1 KB
  __shared__ _Float16 BsT[128][40];  // 10 KB (pad 40 halves = 80 B rows)
  __shared__ float    Cs[16][128];   //  8 KB
  const int tid = threadIdx.x;
  const int lane = tid & 31, wave = tid >> 5;
  const int r = lane & 15, half = lane >> 4;
  v8f c = {};
  for (int k0 = 0; k0 < K; k0 += 32) {
    // ---- stage A (threads 0..127) ----
    if (tid < 128) {
      const int row = tid >> 3, c4 = (tid & 7) * 4;
      const float4 v =
          *reinterpret_cast<const float4*>(A + (size_t)(m0 + row) * K + k0 + c4);
      _Float16* dst = &As[row][c4];
      dst[0] = (_Float16)v.x; dst[1] = (_Float16)v.y;
      dst[2] = (_Float16)v.z; dst[3] = (_Float16)v.w;
    }
    // ---- stage B with transpose (all threads, 4 float4 each) ----
#pragma unroll
    for (int qq = 0; qq < 4; ++qq) {
      const int idx = tid + 256 * qq;  // 0..1023 over 32 rows x 32 float4s
      const int krow = idx >> 5, c4 = (idx & 31) * 4;
      const float4 v = *reinterpret_cast<const float4*>(
          B + (size_t)(k0 + krow) * HW + nblk + c4);
      BsT[c4 + 0][krow] = (_Float16)v.x;
      BsT[c4 + 1][krow] = (_Float16)v.y;
      BsT[c4 + 2][krow] = (_Float16)v.z;
      BsT[c4 + 3][krow] = (_Float16)v.w;
    }
    // ---- prefetch next K chunk of B (global_prefetch_b8) ----
    if (k0 + 32 < K) {
      const int krow = tid >> 3;
      __builtin_prefetch(
          B + (size_t)(k0 + 32 + krow) * HW + nblk + (tid & 7) * 16, 0, 1);
    }
    __syncthreads();
    // ---- fragments: two aligned 16-byte LDS reads per operand ----
    v16h a, b;
    {
      const v8h alo = *reinterpret_cast<const v8h*>(&As[r][8 * half]);
      const v8h ahi = *reinterpret_cast<const v8h*>(&As[r][16 + 8 * half]);
      const _Float16* bp = &BsT[wave * 16 + r][0];
      const v8h blo = *reinterpret_cast<const v8h*>(bp + 8 * half);
      const v8h bhi = *reinterpret_cast<const v8h*>(bp + 16 + 8 * half);
#pragma unroll
      for (int j = 0; j < 8; ++j) {
        a[j] = alo[j]; a[8 + j] = ahi[j];
        b[j] = blo[j]; b[8 + j] = bhi[j];
      }
    }
    c = __builtin_amdgcn_wmma_f32_16x16x32_f16(false, a, false, b, (short)0, c,
                                               false, false);
    __syncthreads();
  }
  // ---- C: stage to LDS, store coalesced b128 ----
#pragma unroll
  for (int j = 0; j < 8; ++j)
    Cs[j + 8 * half][wave * 16 + r] = c[j];
  __syncthreads();
#pragma unroll
  for (int rep = 0; rep < 2; ++rep) {
    const int idx = tid + 256 * rep;  // 0..511 over 16 rows x 32 float4s
    const int row = idx >> 5, c4 = (idx & 31) * 4;
    float4 v;
    v.x = Cs[row][c4 + 0]; v.y = Cs[row][c4 + 1];
    v.z = Cs[row][c4 + 2]; v.w = Cs[row][c4 + 3];
    *reinterpret_cast<float4*>(C + (size_t)(m0 + row) * HW + nblk + c4) = v;
  }
}

// ---------------- K3: q/k 1x1 convs as WMMA GEMMs --------------------------
__global__ __launch_bounds__(256) void gemm_qk_kernel(
    const float* __restrict__ qw, const float* __restrict__ kw,
    const float* __restrict__ f1p, const int* __restrict__ ii,
    const int* __restrict__ jj, float* __restrict__ qbuf,
    float* __restrict__ kbuf) {
  const int nblk = blockIdx.x * 128;
  const int m0 = blockIdx.y * 16;
  const int e = blockIdx.z >> 1, mode = blockIdx.z & 1;
  const float* A = mode ? kw : qw;
  const int img = mode ? jj[e] : ii[e];
  const float* B = f1p + (size_t)img * C3 * HW;
  float* C = (mode ? kbuf : qbuf) + (size_t)e * C6 * HW;
  gemm_tile_block(A, B, C, C3, m0, nblk);
}

// ---------------- K5a: pw 1x1 conv as WMMA GEMM ----------------------------
__global__ __launch_bounds__(256) void gemm_pw_kernel(
    const float* __restrict__ pw, const float* __restrict__ obuf,
    float* __restrict__ pwbuf) {
  const int nblk = blockIdx.x * 128;
  const int m0 = blockIdx.y * 16;
  const int e = blockIdx.z;
  gemm_tile_block(pw, obuf + (size_t)e * C6 * HW,
                  pwbuf + (size_t)e * C4 * HW, C6, m0, nblk);
}

// ---------------- K4: fft + cross-attention + ifft (fused) -----------------
// One 64-thread block (2 full waves -> EXEC all-ones) per (edge, channel).
// Attention GEMMs on WMMA with K padded in LDS; row-contiguous fragment
// gathers are 16-byte ds loads.
__device__ void attn_channel(const _Float16 (*X)[32], const _Float16 (*Y)[32],
                             float (*S)[64], _Float16 (*Pm)[64],
                             float (*AO)[16], float* out12, int t, int lane,
                             int wave) {
  const int r = lane & 15, half = lane >> 4;
  // S = X @ Y^T : 4x4 tiles of 16x16, single K=32 step (cols 12..31 are 0).
#pragma unroll
  for (int tt = 0; tt < 8; ++tt) {
    const int ti = wave * 2 + (tt >> 2);
    const int tj = tt & 3;
    v16h a, b;
    {
      const v8h alo = *reinterpret_cast<const v8h*>(&X[ti * 16 + r][8 * half]);
      const v8h ahi =
          *reinterpret_cast<const v8h*>(&X[ti * 16 + r][16 + 8 * half]);
      const v8h blo = *reinterpret_cast<const v8h*>(&Y[tj * 16 + r][8 * half]);
      const v8h bhi =
          *reinterpret_cast<const v8h*>(&Y[tj * 16 + r][16 + 8 * half]);
#pragma unroll
      for (int j = 0; j < 8; ++j) {
        a[j] = alo[j]; a[8 + j] = ahi[j];
        b[j] = blo[j]; b[8 + j] = bhi[j];  // B[k][n] = Y[n][k]
      }
    }
    v8f cf = {};
    cf = __builtin_amdgcn_wmma_f32_16x16x32_f16(false, a, false, b, (short)0,
                                                cf, false, false);
#pragma unroll
    for (int j = 0; j < 8; ++j)
      S[ti * 16 + j + 8 * half][tj * 16 + r] = cf[j];
  }
  __syncthreads();
  // Row softmax (scale then max-sub); store unnormalized exp as f16,
  // defer 1/sum until after the P@V matmul.
  const float scale = 0.28867513459481287f;  // 1/sqrt(12)
  float mx = -3.0e38f;
  for (int j = 0; j < 64; ++j) mx = fmaxf(mx, S[t][j] * scale);
  float sum = 0.f;
  for (int j = 0; j < 64; ++j) {
    const float p = __expf(S[t][j] * scale - mx);
    Pm[t][j] = (_Float16)p;
    sum += p;
  }
  const float rinv = 1.0f / sum;
  __syncthreads();
  // AO = P @ Y : 4x1 tiles, K=64 in two 32-steps.
#pragma unroll
  for (int w2 = 0; w2 < 2; ++w2) {
    const int ti = wave * 2 + w2;
    v8f cf = {};
#pragma unroll
    for (int k0 = 0; k0 < 64; k0 += 32) {
      v16h a, b;
      const v8h alo =
          *reinterpret_cast<const v8h*>(&Pm[ti * 16 + r][k0 + 8 * half]);
      const v8h ahi =
          *reinterpret_cast<const v8h*>(&Pm[ti * 16 + r][k0 + 16 + 8 * half]);
#pragma unroll
      for (int j = 0; j < 8; ++j) {
        a[j] = alo[j]; a[8 + j] = ahi[j];
      }
#pragma unroll
      for (int j = 0; j < 8; ++j) {  // V gather (column access, strided)
        const int kb = ((j < 4) ? (2 * j) : (16 + 2 * (j - 4))) + 8 * half;
        b[2 * j]     = Y[k0 + kb][r];
        b[2 * j + 1] = Y[k0 + kb + 1][r];
      }
      cf = __builtin_amdgcn_wmma_f32_16x16x32_f16(false, a, false, b, (short)0,
                                                  cf, false, false);
    }
#pragma unroll
    for (int j = 0; j < 8; ++j)
      AO[ti * 16 + j + 8 * half][r] = cf[j];
  }
  __syncthreads();
#pragma unroll
  for (int d = 0; d < 12; ++d) out12[d] = AO[t][d] * rinv;
  __syncthreads();  // allow S/Pm/AO reuse by next call
}

__global__ __launch_bounds__(64) void fft_attn_kernel(
    const float* qbuf, const float* __restrict__ kbuf,
    const float* __restrict__ d8, const int* __restrict__ jj, float* obuf) {
  const int c = blockIdx.x, e = blockIdx.y;
  const int t = threadIdx.x;                  // token 0..63
  const int lane = threadIdx.x & 31, wave = threadIdx.x >> 5;
  const int ph = t >> 3, pwd = t & 7;

  __shared__ _Float16 Xa[64][32], Xp[64][32], Ya[64][32], Yp[64][32];
  __shared__ float Ssm[64][64];
  __shared__ _Float16 Pm[64][64];
  __shared__ float AO[64][16];

  // -------- load 4x4 patches (128-bit loads), +1e-7 pre-fft bias ----------
  const size_t pbase =
      ((size_t)e * C6 + c) * HW + (size_t)(ph * 4) * 32 + pwd * 4;
  float xq[16], xk[16], xd[16];
  {
    const int img = jj[e];
    const size_t dbase =
        ((size_t)img * C6 + c) * HW + (size_t)(ph * 4) * 32 + pwd * 4;
#pragma unroll
    for (int a = 0; a < 4; ++a) {
      const float4 q4 = *reinterpret_cast<const float4*>(qbuf + pbase + a * 32);
      const float4 k4 = *reinterpret_cast<const float4*>(kbuf + pbase + a * 32);
      const float4 d4 = *reinterpret_cast<const float4*>(d8 + dbase + a * 32);
      xq[a * 4 + 0] = q4.x + 1e-7f; xq[a * 4 + 1] = q4.y + 1e-7f;
      xq[a * 4 + 2] = q4.z + 1e-7f; xq[a * 4 + 3] = q4.w + 1e-7f;
      xk[a * 4 + 0] = k4.x + 1e-7f; xk[a * 4 + 1] = k4.y + 1e-7f;
      xk[a * 4 + 2] = k4.z + 1e-7f; xk[a * 4 + 3] = k4.w + 1e-7f;
      xd[a * 4 + 0] = d4.x + 1e-7f; xd[a * 4 + 1] = d4.y + 1e-7f;
      xd[a * 4 + 2] = d4.z + 1e-7f; xd[a * 4 + 3] = d4.w + 1e-7f;
    }
  }

  // -------- rfft2 (4x4 -> 4x3 complex), twiddles in {1,0,-1,0} ------------
  float qR[12], qI[12], kR[12], kI[12], dR[12], dI[12];
#pragma unroll
  for (int r = 0; r < 4; ++r) {
#pragma unroll
    for (int s = 0; s < 3; ++s) {
      float reQ = 0.f, imQ = 0.f, reK = 0.f, imK = 0.f, reD = 0.f, imD = 0.f;
#pragma unroll
      for (int a = 0; a < 4; ++a) {
#pragma unroll
        for (int b = 0; b < 4; ++b) {
          const int ix = (r * a + s * b) & 3;
          const float cc = (ix == 0) ? 1.f : ((ix == 2) ? -1.f : 0.f);
          const float ss = (ix == 1) ? 1.f : ((ix == 3) ? -1.f : 0.f);
          const float vq = xq[a * 4 + b], vk = xk[a * 4 + b],
                      vd = xd[a * 4 + b];
          reQ += vq * cc; imQ -= vq * ss;
          reK += vk * cc; imK -= vk * ss;
          reD += vd * cc; imD -= vd * ss;
        }
      }
      const int dd = r * 3 + s;
      qR[dd] = reQ; qI[dd] = imQ;
      kR[dd] = reK; kI[dd] = imK;
      dR[dd] = reD; dI[dd] = imD;
    }
  }

  // -------- amp/phase into LDS (K padded 12 -> 32 with zeros) -------------
#pragma unroll
  for (int d = 0; d < 12; ++d) {
    Xa[t][d] = (_Float16)sqrtf(kR[d] * kR[d] + kI[d] * kI[d]);
    Xp[t][d] = (_Float16)atan2f(kI[d], kR[d]);
    Ya[t][d] = (_Float16)sqrtf(dR[d] * dR[d] + dI[d] * dI[d]);
    Yp[t][d] = (_Float16)atan2f(dI[d], dR[d]);
  }
#pragma unroll
  for (int d = 12; d < 32; ++d) {
    Xa[t][d] = (_Float16)0.f; Xp[t][d] = (_Float16)0.f;
    Ya[t][d] = (_Float16)0.f; Yp[t][d] = (_Float16)0.f;
  }
  __syncthreads();

  float kamp[12], kpha[12];
  attn_channel(Xa, Ya, Ssm, Pm, AO, kamp, t, lane, wave);
  attn_channel(Xp, Yp, Ssm, Pm, AO, kpha, t, lane, wave);

  // -------- kc reconstruction, qf*kc, epsilons per reference --------------
  float pr[12], pi[12];
#pragma unroll
  for (int d = 0; d < 12; ++d) {
    const float re = kamp[d] * cosf(kpha[d]) + 2e-7f;  // +1e-7 (real) +1e-7 (kc)
    const float im = kamp[d] * sinf(kpha[d]) + 1e-7f;
    pr[d] = qR[d] * re - qI[d] * im + 1e-7f;           // + 1e-7 on product
    pi[d] = qR[d] * im + qI[d] * re;
  }

  // -------- irfft2 (Hermitian-extend last axis, 4x4 inverse DFT) ----------
  float Gr[4][4], Gi[4][4];
#pragma unroll
  for (int r = 0; r < 4; ++r) {
#pragma unroll
    for (int s = 0; s < 3; ++s) { Gr[r][s] = pr[r * 3 + s]; Gi[r][s] = pi[r * 3 + s]; }
    const int rr = (4 - r) & 3;
    Gr[r][3] = pr[rr * 3 + 1];
    Gi[r][3] = -pi[rr * 3 + 1];
  }
#pragma unroll
  for (int a = 0; a < 4; ++a) {
    float4 orow;
    float* op = &orow.x;
#pragma unroll
    for (int b = 0; b < 4; ++b) {
      float acc = 0.f;
#pragma unroll
      for (int r = 0; r < 4; ++r) {
#pragma unroll
        for (int s = 0; s < 4; ++s) {
          const int ix = (r * a + s * b) & 3;
          const float cc = (ix == 0) ? 1.f : ((ix == 2) ? -1.f : 0.f);
          const float ss = (ix == 1) ? 1.f : ((ix == 3) ? -1.f : 0.f);
          acc += Gr[r][s] * cc - Gi[r][s] * ss;
        }
      }
      op[b] = acc * (1.0f / 16.0f);
    }
    *reinterpret_cast<float4*>(obuf + pbase + a * 32) = orow;  // in-place over q
  }
}

// ---------------- K6: PixelShuffle(r=2) + channel LayerNorm ----------------
__global__ __launch_bounds__(256) void ps_ln_kernel(
    const float* __restrict__ pwbuf, const float* __restrict__ lnw,
    const float* __restrict__ lnb, float* __restrict__ out) {
  const int idx = blockIdx.x * 256 + threadIdx.x;  // e*4096 + y*64 + x
  if (idx >= NE * 64 * 64) return;
  const int x = idx & 63, y = (idx >> 6) & 63, e = idx >> 12;
  const int r = (y & 1) * 2 + (x & 1);
  const int y2 = y >> 1, x2 = x >> 1;
  const float* src = pwbuf + (size_t)e * C4 * HW + (size_t)r * HW + y2 * 32 + x2;
  float vals[64];
  float mu = 0.f;
#pragma unroll
  for (int co = 0; co < 64; ++co) {
    vals[co] = src[(size_t)co * 4 * HW];
    mu += vals[co];
  }
  mu *= (1.f / 64.f);
  float var = 0.f;
#pragma unroll
  for (int co = 0; co < 64; ++co) {
    const float dlt = vals[co] - mu;
    var += dlt * dlt;
  }
  var *= (1.f / 64.f);
  const float rs = rsqrtf(var + 1e-5f);
#pragma unroll
  for (int co = 0; co < 64; ++co)
    out[(size_t)((e * 64 + co) * 64 + y) * 64 + x] =
        (vals[co] - mu) * rs * lnw[co] + lnb[co];
}

// ---------------------------------------------------------------------------
extern "C" void kernel_launch(void* const* d_in, const int* in_sizes, int n_in,
                              void* d_out, int out_size, void* d_ws,
                              size_t ws_size, hipStream_t stream) {
  const float* f1   = (const float*)d_in[0];
  const float* de   = (const float*)d_in[1];
  const int*   ii   = (const int*)d_in[2];
  const int*   jj   = (const int*)d_in[3];
  const float* pesw = (const float*)d_in[4];
  const float* pesb = (const float*)d_in[5];
  const float* pemw = (const float*)d_in[6];
  const float* pemb = (const float*)d_in[7];
  const float* pelw = (const float*)d_in[8];
  const float* pelb = (const float*)d_in[9];
  const float* edw  = (const float*)d_in[10];
  const float* edb  = (const float*)d_in[11];
  const float* bng  = (const float*)d_in[12];
  const float* bnb  = (const float*)d_in[13];
  const float* bnrm = (const float*)d_in[14];
  const float* bnrv = (const float*)d_in[15];
  const float* qw   = (const float*)d_in[16];
  const float* kw   = (const float*)d_in[17];
  const float* pww  = (const float*)d_in[18];
  const float* lnw  = (const float*)d_in[19];
  const float* lnb  = (const float*)d_in[20];
  (void)in_sizes; (void)n_in; (void)out_size; (void)ws_size;

  char* ws = (char*)d_ws;
  const size_t SZ_F1P = sizeof(float) * (size_t)NIMG * C3 * HW;  //  6.0 MB
  const size_t SZ_D8  = sizeof(float) * (size_t)NIMG * C6 * HW;  // 12.0 MB
  const size_t SZ_QK  = sizeof(float) * (size_t)NE * C6 * HW;    // 48.0 MB
  float* f1p   = (float*)(ws);
  float* d8    = (float*)(ws + SZ_F1P);
  float* qbuf  = (float*)(ws + SZ_F1P + SZ_D8);
  float* kbuf  = (float*)(ws + SZ_F1P + SZ_D8 + SZ_QK);
  float* pwbuf = (float*)(ws + SZ_F1P + SZ_D8 + 2 * SZ_QK);
  float* obuf  = qbuf;  // in-place: each K4 thread reads its q patch first

  patch_embed_kernel<<<dim3((NIMG * C3 * HW) / 256), dim3(256), 0, stream>>>(
      f1, pesw, pesb, pemw, pemb, pelw, pelb, f1p);
  embedd_kernel<<<dim3((NIMG * C6 * HW) / 256), dim3(256), 0, stream>>>(
      de, edw, edb, bng, bnb, bnrm, bnrv, d8);
  gemm_qk_kernel<<<dim3(8, C6 / 16, NE * 2), dim3(256), 0, stream>>>(
      qw, kw, f1p, ii, jj, qbuf, kbuf);
  fft_attn_kernel<<<dim3(C6, NE), dim3(64), 0, stream>>>(qbuf, kbuf, d8, jj,
                                                         obuf);
  gemm_pw_kernel<<<dim3(8, C4 / 16, NE), dim3(256), 0, stream>>>(pww, obuf,
                                                                 pwbuf);
  ps_ln_kernel<<<dim3((NE * 64 * 64) / 256), dim3(256), 0, stream>>>(
      pwbuf, lnw, lnb, (float*)d_out);
}
